// SoftAlignAttentionMixed_23373212025422
// MI455X (gfx1250) — matrix operand order repaired and checked by reading, not compiled, not run on hardware
//
#include <hip/hip_runtime.h>
#include <math.h>
#include <stdint.h>

#define B_  4
#define C_  512
#define T_  2048
#define H_  8
#define W_  16
#define TS_ 1000
#define D_  64
#define SCALE_ 0.125f   // D^-0.5

typedef __attribute__((ext_vector_type(16))) _Float16 v16h;
typedef __attribute__((ext_vector_type(8)))  float    v8f;
typedef __attribute__((ext_vector_type(4)))  unsigned int v4u;
typedef __attribute__((ext_vector_type(8)))  int      v8i;
typedef __attribute__((ext_vector_type(4)))  int      v4i;

#define WMMA(a, b, c) __builtin_amdgcn_wmma_f32_16x16x32_f16(false,(a),false,(b),(short)0,(c),false,false)

// ---------------- CDNA5 16x16x32 f16 fragment loaders ----------------
// A-matrix 16x32 (MxK): lanes 0-15 k in {0..7,16..23}, lanes 16-31 +8 (ISA 7.12.2).
// Tile in LDS row-major [m][k], k contiguous.
__device__ __forceinline__ v16h load_fragA(const _Float16* base, int ld) {
  const int l = threadIdx.x & 31;
  const int m = l & 15;
  const int koff = (l >> 4) << 3;
  const _Float16* p = base + m * ld + koff;
  v16h f;
  #pragma unroll
  for (int j = 0; j < 8; ++j) {
    const int k = ((j >> 2) << 4) + ((j & 3) << 1);
    f[2*j]   = p[k];
    f[2*j+1] = p[k+1];
  }
  return f;
}

// B-matrix 32x16 (KxN): lane holds column n=L%16; lanes 0-15 k in {0..7,16..23},
// lanes 16-31 +8 (matches A). Tile in LDS as [n][k], k contiguous.
__device__ __forceinline__ v16h load_fragB(const _Float16* base, int ld) {
  const int l = threadIdx.x & 31;
  const int n = l & 15;
  const int koff = (l >> 4) << 3;
  const _Float16* p = base + n * ld + koff;
  v16h f;
  #pragma unroll
  for (int j = 0; j < 16; ++j) f[j] = p[(j < 8) ? j : j + 8];
  return f;
}

// ---------------- TDM: 2D f16 tile, global -> LDS (D# per ISA 8.3-8.5) ----------
// tile_d0 = contiguous elements per row, tile_d1 = rows, stride_d0 = row stride
// (elements). pad_code: LDS pad of 1 DWORD every 2^(pad_code+1) DWORDs.
__device__ __forceinline__ void tdm_load_2d(unsigned lds_byte_addr, const void* gaddr,
                                            unsigned tile_d0, unsigned tile_d1,
                                            unsigned stride_d0, unsigned pad_code)
{
  const unsigned long long ga = (unsigned long long)(uintptr_t)gaddr;
  v4u g0;
  g0[0] = 1u;                                             // count=1, user mode
  g0[1] = lds_byte_addr;                                  // LDS dest (bytes)
  g0[2] = (unsigned)(ga & 0xffffffffu);                   // global_addr[31:0]
  g0[3] = (unsigned)((ga >> 32) & 0x1ffffffu) | (2u << 30); // ga[56:32] | type=2
  v8i g1;
  g1[0] = (int)((1u << 16) | (1u << 20) | (pad_code << 22)); // 2B elems, pad_en, amt=1dw
  g1[1] = (int)(tile_d0 << 16);                           // tensor_dim0 = tile_d0
  g1[2] = (int)(tile_d1 << 16);                           // tensor_dim1 = tile_d1
  g1[3] = (int)(tile_d0 << 16);                           // tile_dim0
  g1[4] = (int)tile_d1;                                   // tile_dim1 (tile_dim2=0)
  g1[5] = (int)stride_d0;                                 // tensor_dim0_stride
  g1[6] = 0;
  g1[7] = 0;
  const v4i z4 = {0, 0, 0, 0};
  const v8i z8 = {0, 0, 0, 0, 0, 0, 0, 0};
  __builtin_amdgcn_tensor_load_to_lds(g0, g1, z4, z4, z8, 0);
}

// ---------------- per-batch mixing scales ----------------
__global__ void scales_kernel(const int* __restrict__ t, float* __restrict__ sg,
                              float* __restrict__ sl) {
  const int i = threadIdx.x;
  if (i < B_) {
    const float tn = (float)t[i] / (float)(TS_ - 1);
    sg[i] = sqrtf(1.f - tn);
    sl[i] = sqrtf(tn);
  }
}

// ---------------- projection GEMM: r[o,t] = sum_c W[o,c] X[b,c,t] + bias[o]
// 64x64 block tile, 4 waves, 2x2 fragments per wave.
// mode 0: Y=r (f32 planar)        mode 1: Y=scale[b]*r (f32 planar)
// mode 2: Y+=scale[b]*r           mode 3: Y16=r (f16 [b][h][t][d])
// mode 4: Y16=r (f16 planar [b][c][t])
__global__ __launch_bounds__(128) void proj_gemm_kernel(
    const float* __restrict__ Wt, const float* __restrict__ bias,
    const float* __restrict__ X, float* __restrict__ Y, _Float16* __restrict__ Y16,
    const float* __restrict__ scale, int mode)
{
  const int tid = threadIdx.x;
  const int b  = blockIdx.z;
  const int m0 = blockIdx.y * 64;
  const int n0 = blockIdx.x * 64;
  X += (size_t)b * C_ * T_;
  __shared__ _Float16 As[64][34];   // [m][k]
  __shared__ _Float16 Bs[64][34];   // [n][k]
  const int wid = tid >> 5;
  const int mw = (wid >> 1) << 5;
  const int nw = (wid & 1) << 5;
  v8f a00 = {}, a01 = {}, a10 = {}, a11 = {};
  for (int k0 = 0; k0 < C_; k0 += 32) {
    __syncthreads();
    {
      const int r = tid >> 1, c0 = (tid & 1) << 4;
      const float* src = Wt + (size_t)(m0 + r) * C_ + k0 + c0;
      #pragma unroll
      for (int i = 0; i < 16; ++i) As[r][c0 + i] = (_Float16)src[i];
    }
    {
      const int kk = tid >> 2, nn0 = (tid & 3) << 4;
      const float* src = X + (size_t)(k0 + kk) * T_ + n0 + nn0;
      #pragma unroll
      for (int i = 0; i < 16; ++i) Bs[nn0 + i][kk] = (_Float16)src[i];
    }
    __syncthreads();
    v16h fa0 = load_fragA(&As[mw][0],      34);
    v16h fa1 = load_fragA(&As[mw + 16][0], 34);
    v16h fb0 = load_fragB(&Bs[nw][0],      34);
    v16h fb1 = load_fragB(&Bs[nw + 16][0], 34);
    a00 = WMMA(fa0, fb0, a00); a01 = WMMA(fa0, fb1, a01);
    a10 = WMMA(fa1, fb0, a10); a11 = WMMA(fa1, fb1, a11);
  }
  const int l = tid & 31, cl = l & 15, ro = (l >> 4) << 3;
  const size_t bofs = (size_t)b * C_ * T_;
  auto store = [&](const v8f& a, int rbase, int cbase) {
    #pragma unroll
    for (int v = 0; v < 8; ++v) {
      const int row = rbase + ro + v;
      const int col = cbase + cl;
      const float val = a[v] + bias[row];
      if (mode == 0)      Y[bofs + (size_t)row * T_ + col] = val;
      else if (mode == 1) Y[bofs + (size_t)row * T_ + col] = scale[b] * val;
      else if (mode == 2) Y[bofs + (size_t)row * T_ + col] += scale[b] * val;
      else if (mode == 3) {
        const int h = row >> 6, d = row & 63;
        Y16[bofs + ((size_t)h * T_ + col) * D_ + d] = (_Float16)val;
      } else              Y16[bofs + (size_t)row * T_ + col] = (_Float16)val;
    }
  };
  store(a00, m0 + mw,      n0 + nw);
  store(a01, m0 + mw,      n0 + nw + 16);
  store(a10, m0 + mw + 16, n0 + nw);
  store(a11, m0 + mw + 16, n0 + nw + 16);
}

// ---------------- RoPE on f16 [b*h][t][d] (pairs adjacent in d) ----------------
__global__ __launch_bounds__(256) void rope16_kernel(_Float16* __restrict__ q,
                                                     _Float16* __restrict__ k) {
  const int idx = blockIdx.x * 256 + threadIdx.x;   // (bh*T + t)*32 + i
  const int i  = idx & (D_/2 - 1);
  const int t  = (idx >> 5) & (T_ - 1);
  if ((idx >> 16) >= B_ * H_) return;
  const float inv_freq = __expf(-(2.f * (float)i / (float)D_) * logf(10000.f));
  const float ang = (float)t * inv_freq;
  const float c = cosf(ang), s = sinf(ang);
  const size_t a1 = ((size_t)(idx >> 5) << 6) + 2 * i;
  float x1 = (float)q[a1], x2 = (float)q[a1 + 1];
  q[a1]     = (_Float16)(x1 * c - x2 * s);
  q[a1 + 1] = (_Float16)(x1 * s + x2 * c);
  x1 = (float)k[a1]; x2 = (float)k[a1 + 1];
  k[a1]     = (_Float16)(x1 * c - x2 * s);
  k[a1 + 1] = (_Float16)(x1 * s + x2 * c);
}

// ---------------- local windowed attention (tiny FLOPs -> VALU) ----------------
__global__ __launch_bounds__(128) void local_attn_kernel(
    const float* __restrict__ q, const float* __restrict__ k,
    const float* __restrict__ v, float* __restrict__ out)
{
  const int idx = blockIdx.x * 128 + threadIdx.x;
  const int t  = idx & (T_ - 1);
  const int bh = idx >> 11;
  if (bh >= B_ * H_) return;
  const size_t base = (size_t)bh * D_ * T_;
  float qv[D_];
  #pragma unroll
  for (int d = 0; d < D_; ++d) qv[d] = q[base + (size_t)d * T_ + t];
  float dots[W_];
  #pragma unroll
  for (int w = 0; w < W_; ++w) {
    const int tk = t + w - (W_/2);
    float acc = 0.f;
    if (tk >= 0 && tk < T_) {
      #pragma unroll
      for (int d = 0; d < D_; ++d) acc += qv[d] * k[base + (size_t)d * T_ + tk];
    }
    dots[w] = acc * SCALE_;   // zero-padded logits, matching reference padding
  }
  float mx = dots[0];
  #pragma unroll
  for (int w = 1; w < W_; ++w) mx = fmaxf(mx, dots[w]);
  float s = 0.f;
  #pragma unroll
  for (int w = 0; w < W_; ++w) { dots[w] = __expf(dots[w] - mx); s += dots[w]; }
  const float inv = 1.f / s;
  float ov[D_];
  #pragma unroll
  for (int d = 0; d < D_; ++d) ov[d] = 0.f;
  #pragma unroll
  for (int w = 0; w < W_; ++w) {
    const int tk = t + w - (W_/2);
    if (tk >= 0 && tk < T_) {
      const float aw = dots[w] * inv;
      #pragma unroll
      for (int d = 0; d < D_; ++d) ov[d] += aw * v[base + (size_t)d * T_ + tk];
    }
  }
  #pragma unroll
  for (int d = 0; d < D_; ++d) out[base + (size_t)d * T_ + t] = ov[d];
}

// ---------------- global attention: flash + WMMA + TDM double-buffered K/V ------
// Qg/Kg: f16 [b*h][t][d];  Vg: f16 [b*h][d][t];  Og: f32 planar [b*h][d][t]
__global__ __launch_bounds__(128) void flash_kernel(
    const _Float16* __restrict__ Qg, const _Float16* __restrict__ Kg,
    const _Float16* __restrict__ Vg, float* __restrict__ Og)
{
  const int tid = threadIdx.x;
  const int bh  = blockIdx.y;
  const int t0  = blockIdx.x * 64;
  __shared__ _Float16 Qs[64][66];        // [m=t_local][k=d]
  __shared__ _Float16 Ks[2][32][66];     // [n=t_col][k=d], double buffered
  __shared__ _Float16 Vs[2][64][34];     // [n=d][k=t_col], double buffered
  __shared__ _Float16 Ps[4][16][34];     // per-wave P bounce (C-layout -> A-layout)
  const _Float16* qbase = Qg + ((size_t)bh * T_ + t0) * D_;
  const _Float16* kbase = Kg + (size_t)bh * T_ * D_;
  const _Float16* vbase = Vg + (size_t)bh * D_ * T_;
  if (tid < 32) {   // wave 0 drives the Tensor Data Mover
    tdm_load_2d((unsigned)(uintptr_t)&Qs[0][0],    qbase, 64, 64, 64, 4);
    tdm_load_2d((unsigned)(uintptr_t)&Ks[0][0][0], kbase, 64, 32, 64, 4);
    tdm_load_2d((unsigned)(uintptr_t)&Vs[0][0][0], vbase, 32, 64, T_, 3);
    __builtin_amdgcn_s_wait_tensorcnt(0);
  }
  __syncthreads();

  const int wid = tid >> 5;
  const int l = tid & 31, cl = l & 15, ro = (l >> 4) << 3;
  const int mb = wid * 16;
  v8f o0 = {}, o1 = {}, o2 = {}, o3 = {};
  float mrow[8], lrow[8];
  #pragma unroll
  for (int v = 0; v < 8; ++v) { mrow[v] = -1e30f; lrow[v] = 0.f; }

  for (int j = 0; j < T_; j += 32) {
    const int cur = (j >> 5) & 1, nxt = cur ^ 1;
    if (tid < 32 && (j + 32) < T_) {   // prefetch next tile while computing
      tdm_load_2d((unsigned)(uintptr_t)&Ks[nxt][0][0],
                  kbase + (size_t)(j + 32) * D_, 64, 32, 64, 4);
      tdm_load_2d((unsigned)(uintptr_t)&Vs[nxt][0][0],
                  vbase + (j + 32),              32, 64, T_, 3);
    }
    // S = Q(16x64) . K^T(64x32): two 16-col tiles
    v8f s0 = {}, s1 = {};
    {
      v16h fa0 = load_fragA(&Qs[mb][0],  66);
      v16h fa1 = load_fragA(&Qs[mb][32], 66);
      v16h b00 = load_fragB(&Ks[cur][0][0],   66);
      v16h b01 = load_fragB(&Ks[cur][0][32],  66);
      v16h b10 = load_fragB(&Ks[cur][16][0],  66);
      v16h b11 = load_fragB(&Ks[cur][16][32], 66);
      s0 = WMMA(fa0, b00, s0); s0 = WMMA(fa1, b01, s0);
      s1 = WMMA(fa0, b10, s1); s1 = WMMA(fa1, b11, s1);
    }
    // online softmax; C-layout: element v -> row ro+v, col cl; row spans 16 lanes
    #pragma unroll
    for (int v = 0; v < 8; ++v) {
      const float x0 = s0[v] * SCALE_, x1 = s1[v] * SCALE_;
      float mx = fmaxf(x0, x1);
      #pragma unroll
      for (int off = 8; off; off >>= 1) mx = fmaxf(mx, __shfl_xor(mx, off, 32));
      const float mn = fmaxf(mrow[v], mx);
      const float al = __expf(mrow[v] - mn);
      mrow[v] = mn;
      const float p0 = __expf(x0 - mn), p1 = __expf(x1 - mn);
      float rs = p0 + p1;
      #pragma unroll
      for (int off = 8; off; off >>= 1) rs += __shfl_xor(rs, off, 32);
      lrow[v] = lrow[v] * al + rs;
      o0[v] *= al; o1[v] *= al; o2[v] *= al; o3[v] *= al;
      Ps[wid][ro + v][cl]      = (_Float16)p0;
      Ps[wid][ro + v][16 + cl] = (_Float16)p1;
    }
    // O += P(16x32) . V(32x64)
    v16h a2  = load_fragA(&Ps[wid][0][0], 34);
    v16h bv0 = load_fragB(&Vs[cur][0][0],  34);
    v16h bv1 = load_fragB(&Vs[cur][16][0], 34);
    v16h bv2 = load_fragB(&Vs[cur][32][0], 34);
    v16h bv3 = load_fragB(&Vs[cur][48][0], 34);
    o0 = WMMA(a2, bv0, o0); o1 = WMMA(a2, bv1, o1);
    o2 = WMMA(a2, bv2, o2); o3 = WMMA(a2, bv3, o3);
    if (tid < 32) __builtin_amdgcn_s_wait_tensorcnt(0);
    __syncthreads();
  }
  const size_t obase = (size_t)bh * D_ * T_;
  #pragma unroll
  for (int v = 0; v < 8; ++v) {
    const float inv = 1.f / lrow[v];
    const int trow = t0 + mb + ro + v;
    Og[obase + (size_t)(cl)      * T_ + trow] = o0[v] * inv;
    Og[obase + (size_t)(16 + cl) * T_ + trow] = o1[v] * inv;
    Og[obase + (size_t)(32 + cl) * T_ + trow] = o2[v] * inv;
    Og[obase + (size_t)(48 + cl) * T_ + trow] = o3[v] * inv;
  }
}

extern "C" void kernel_launch(void* const* d_in, const int* in_sizes, int n_in,
                              void* d_out, int out_size, void* d_ws, size_t ws_size,
                              hipStream_t stream) {
  (void)in_sizes; (void)n_in; (void)out_size;
  const float* x    = (const float*)d_in[0];
  const float* cond = (const float*)d_in[1];
  const int*   tt   = (const int*)  d_in[2];
  const float* lq_w = (const float*)d_in[3];
  const float* lq_b = (const float*)d_in[4];
  const float* lk_w = (const float*)d_in[5];
  const float* lk_b = (const float*)d_in[6];
  const float* lv_w = (const float*)d_in[7];
  const float* lv_b = (const float*)d_in[8];
  const float* lo_w = (const float*)d_in[9];
  const float* lo_b = (const float*)d_in[10];
  const float* gq_w = (const float*)d_in[11];
  const float* gq_b = (const float*)d_in[12];
  const float* gk_w = (const float*)d_in[13];
  const float* gk_b = (const float*)d_in[14];
  const float* gv_w = (const float*)d_in[15];
  const float* gv_b = (const float*)d_in[16];
  const float* go_w = (const float*)d_in[17];
  const float* go_b = (const float*)d_in[18];
  float* out = (float*)d_out;

  const size_t planar = (size_t)B_ * C_ * T_;
  const size_t need = (5 * planar + 16) * sizeof(float) + 3 * planar * sizeof(_Float16);
  if (ws_size < need) return;
  float* f = (float*)d_ws;
  float* lq = f + 0*planar; float* lk = f + 1*planar; float* lv = f + 2*planar;
  float* ctxl = f + 3*planar; float* ctxg = f + 4*planar;
  float* sg = f + 5*planar; float* sl = sg + B_;
  _Float16* h0 = (_Float16*)(f + 5*planar + 16);
  _Float16* gq16 = h0;
  _Float16* gk16 = h0 + planar;
  _Float16* gv16 = h0 + 2*planar;

  scales_kernel<<<1, 32, 0, stream>>>(tt, sg, sl);

  dim3 gg(T_/64, C_/64, B_);
  proj_gemm_kernel<<<gg, 128, 0, stream>>>(lq_w, lq_b, x,    lq,  nullptr, nullptr, 0);
  proj_gemm_kernel<<<gg, 128, 0, stream>>>(lk_w, lk_b, cond, lk,  nullptr, nullptr, 0);
  proj_gemm_kernel<<<gg, 128, 0, stream>>>(lv_w, lv_b, cond, lv,  nullptr, nullptr, 0);
  proj_gemm_kernel<<<gg, 128, 0, stream>>>(gq_w, gq_b, x,    nullptr, gq16, nullptr, 3);
  proj_gemm_kernel<<<gg, 128, 0, stream>>>(gk_w, gk_b, cond, nullptr, gk16, nullptr, 3);
  proj_gemm_kernel<<<gg, 128, 0, stream>>>(gv_w, gv_b, cond, nullptr, gv16, nullptr, 4);

  rope16_kernel<<<(B_*H_*(D_/2)*T_)/256, 256, 0, stream>>>(gq16, gk16);
  local_attn_kernel<<<(B_*H_*T_)/128, 128, 0, stream>>>(lq, lk, lv, ctxl);
  flash_kernel<<<dim3(T_/64, B_*H_), 128, 0, stream>>>(gq16, gk16, gv16, ctxg);

  // out = sqrt(a_g)*GoProj(ctxg) + sqrt(a_l)*LoProj(ctxl), mix fused in epilogues
  proj_gemm_kernel<<<gg, 128, 0, stream>>>(go_w, go_b, ctxg, out, nullptr, sg, 1);
  proj_gemm_kernel<<<gg, 128, 0, stream>>>(lo_w, lo_b, ctxl, out, nullptr, sl, 2);
}
